// PredictionHead_70282844832281
// MI455X (gfx1250) — matrix operand order repaired and checked by reading, not compiled, not run on hardware
//
#include <hip/hip_runtime.h>
#include <math.h>

// ---------------------------------------------------------------------------
// Problem constants (from reference)
// ---------------------------------------------------------------------------
#define BQ     8
#define HH     256
#define WW     256
#define F_PIX  128
#define QD     256
#define HID    512
#define QPB    1024
#define QTOT   8192
#define WS7    7
#define NWIN   49            // 7*7
#define NEG_BIAS (-1000000.0f)
#define LOG_MAX_COV 11.512925464970229f
#define EPS_F 1e-06f

// Output offsets (floats) in d_out, concatenated in reference return order
#define O_LOGIT   ((size_t)0)              // [8192]
#define O_ISE     ((size_t)8192)           // [8192]
#define O_BML     ((size_t)16384)          // [8192*49]
#define O_PL      ((size_t)417792)         // [8192*49]
#define O_KIDX    ((size_t)819200)         // [8192*49*3]
#define O_ISPEC   ((size_t)2023424)        // [8192*49]
#define O_CHOL    ((size_t)2424832)        // [8192*4]
#define O_SIDX    ((size_t)2457600)        // [8192*49*4]

// Workspace offsets (bytes)
#define WS_QB      ((size_t)0)                       // queries bf16   8192*256*2 = 4 MiB
#define WS_CLSW1P  (WS_QB     + 4194304)             // 256x512 packed bf16 = 256 KiB
#define WS_QPW1P   (WS_CLSW1P + 262144)
#define WS_SDW1P   (WS_QPW1P  + 262144)
#define WS_PMW1P   (WS_SDW1P  + 262144)              // 128x512 = 128 KiB
#define WS_PMW2P   (WS_PMW1P  + 131072)              // 512x256 = 256 KiB
#define WS_QPW2P   (WS_PMW2P  + 262144)              // 512x128 = 128 KiB
#define WS_HCLS    (WS_QPW2P  + 131072)              // 8192x512 bf16 = 8 MiB
#define WS_HQP     (WS_HCLS   + 8388608)
#define WS_HSD     (WS_HQP    + 8388608)
#define WS_MLPQ    (WS_HSD    + 8388608)             // 8192x128 f32 = 4 MiB

typedef __attribute__((ext_vector_type(16))) __bf16 v16bf;
typedef __attribute__((ext_vector_type(8)))  float  v8f;

__device__ __forceinline__ unsigned short f32_to_bf16(float f) {
    unsigned int u = __float_as_uint(f);
    unsigned int r = (u + 0x7FFFu + ((u >> 16) & 1u)) >> 16;   // RNE
    return (unsigned short)r;
}
__device__ __forceinline__ float bf16_to_f32(unsigned short h) {
    return __uint_as_float(((unsigned int)h) << 16);
}
__device__ __forceinline__ float wave_sum(float s) {
#pragma unroll
    for (int m = 16; m; m >>= 1) s += __shfl_xor(s, m, 32);
    return s;
}

// A-fragment (16x32 bf16, MxK) loaded from LDS, row-major with leading dim ld.
// Lane L: m = L&15, g = L>>4.  k(e) = kc + g*8 + e  (e<8),  kc + 16 + g*8 + (e-8).
__device__ __forceinline__ v16bf load_a_lds(const unsigned short* X, int ld,
                                            int row_base, int kc, int lane) {
    int g = lane >> 4, m = lane & 15;
    const unsigned short* p = X + (size_t)(row_base + m) * ld + kc + g * 8;
    union { v16bf v; uint4 u4[2]; } fa;
    fa.u4[0] = *(const uint4*)(p);
    fa.u4[1] = *(const uint4*)(p + 16);
    return fa.v;
}

// B-fragment (32x16 bf16, KxN) from the packed global layout produced by pack_b:
// per (kt,nt) tile: 32 lanes x 16 contiguous bf16 per lane.
__device__ __forceinline__ v16bf load_b_packed(const unsigned short* WP,
                                               int ntiles_n, int kt, int nt, int lane) {
    const unsigned short* p = WP + (((size_t)kt * ntiles_n + nt) * 32 + lane) * 16;
    union { v16bf v; uint4 u4[2]; } fb;
    const uint4* p4 = (const uint4*)p;
    fb.u4[0] = p4[0];
    fb.u4[1] = p4[1];
    return fb.v;
}

__device__ __forceinline__ v8f wmma_bf16(v16bf a, v16bf b, v8f c) {
    return __builtin_amdgcn_wmma_f32_16x16x32_bf16(false, a, false, b,
                                                   (short)0, c, false, false);
}

// CDNA5 async global -> LDS copy (ASYNCcnt) + wait, per CDNA5 ISA.
// lds_off is the raw LDS byte offset of this lane's destination (staging buffer is
// the kernel's only LDS object, so its base offset is 0).
__device__ __forceinline__ void async_g2l_b128(unsigned lds_off, const void* gptr) {
    unsigned long long ga = (unsigned long long)gptr;
    asm volatile("global_load_async_to_lds_b128 %0, %1, off"
                 :: "v"(lds_off), "v"(ga) : "memory");
}
__device__ __forceinline__ void wait_async0() {
    asm volatile("s_wait_asynccnt 0x0" ::: "memory");
}

// ---------------------------------------------------------------------------
// Elementwise f32 -> bf16 conversion (queries)
// ---------------------------------------------------------------------------
__global__ void cvt_bf16_kernel(const float* __restrict__ src,
                                unsigned short* __restrict__ dst, int n) {
    int i = blockIdx.x * 256 + threadIdx.x;
    if (i < n) dst[i] = f32_to_bf16(src[i]);
}

// ---------------------------------------------------------------------------
// Repack f32 W[K,N] -> WMMA-B-fragment-tiled bf16
//   k: kt = k>>5, g = (k>>3)&1, e = (k&7) + ((k&31)>>4)*8 ; n: nt = n>>4, lane = g*16 + (n&15)
// ---------------------------------------------------------------------------
__global__ void pack_b_kernel(const float* __restrict__ W,
                              unsigned short* __restrict__ WP, int K, int N) {
    int idx = blockIdx.x * 256 + threadIdx.x;
    if (idx >= K * N) return;
    int k = idx / N, n = idx - k * N;
    int kt = k >> 5, kr = k & 31;
    int g = (kr >> 3) & 1, e = (kr & 7) + ((kr >> 4) << 3);
    int nt = n >> 4, nr = n & 15;
    int lane = g * 16 + nr;
    size_t off = (((size_t)kt * (N >> 4) + nt) * 32 + lane) * 16 + e;
    WP[off] = f32_to_bf16(W[idx]);
}

// ---------------------------------------------------------------------------
// Tiled WMMA GEMM:  C[M,N] = act(A[M,K](bf16) @ Bpacked[K,N] + bias)
// Block tile 128x128, 256 threads (8 waves): wave w owns row-tile w and all 8
// col tiles.  A tile double-buffered in LDS, staged with CDNA5 async
// global->LDS b128 copies overlapped with WMMA compute; B fragments are
// clause-loaded from the L2-resident packed weights.
// ---------------------------------------------------------------------------
__global__ void gemm_wmma_kernel(const unsigned short* __restrict__ A,
                                 const unsigned short* __restrict__ BP,
                                 const float* __restrict__ bias,
                                 void* __restrict__ Cout,
                                 int M, int N, int K, int do_relu, int out_f32) {
    __shared__ unsigned short XS[2][128 * 32];   // only LDS object: base offset 0
    int t = threadIdx.x;
    int wave = t >> 5, lane = t & 31;
    int g = lane >> 4, ln = lane & 15;
    int brow = blockIdx.x * 128;
    int bcol = blockIdx.y * 128;
    int ntile_n = N >> 4;
    int bctile = bcol >> 4;

    // each thread stages 2 x 16B per K-chunk (rows r and r+64)
    int sr = t >> 2, scg = (t & 3) << 3;

    auto stage = [&](int buf, int kt) {
        unsigned lds0 = (unsigned)(buf * 8192 + (sr * 32 + scg) * 2);
        const unsigned short* g0 = A + (size_t)(brow + sr) * K + (kt << 5) + scg;
        async_g2l_b128(lds0, g0);
        unsigned lds1 = lds0 + (unsigned)(64 * 32 * 2);
        const unsigned short* g1 = g0 + (size_t)64 * K;
        async_g2l_b128(lds1, g1);
    };

    v8f acc[8];
#pragma unroll
    for (int j = 0; j < 8; ++j) acc[j] = (v8f){0.f,0.f,0.f,0.f,0.f,0.f,0.f,0.f};

    int kts = K >> 5;
    stage(0, 0);
    wait_async0();
    __syncthreads();

    for (int kt = 0; kt < kts; ++kt) {
        int buf = kt & 1;
        if (kt + 1 < kts) stage(buf ^ 1, kt + 1);

        v16bf a = load_a_lds(&XS[buf][0], 32, wave * 16, 0, lane);
        v16bf bf[8];
#pragma unroll
        for (int j = 0; j < 8; ++j)
            bf[j] = load_b_packed(BP, ntile_n, kt, bctile + j, lane);
#pragma unroll
        for (int j = 0; j < 8; ++j)
            acc[j] = wmma_bf16(a, bf[j], acc[j]);

        wait_async0();        // next buffer fully staged (this wave)
        __syncthreads();      // ... and by all waves, before it is read
    }

#pragma unroll
    for (int j = 0; j < 8; ++j) {
        int col = bcol + j * 16 + ln;
        float bv = bias ? bias[col] : 0.f;
#pragma unroll
        for (int i = 0; i < 8; ++i) {
            int row = brow + wave * 16 + g * 8 + i;
            float v = acc[j][i] + bv;
            if (do_relu) v = fmaxf(v, 0.f);
            if (out_f32)
                ((float*)Cout)[(size_t)row * N + col] = v;
            else
                ((unsigned short*)Cout)[(size_t)row * N + col] = f32_to_bf16(v);
        }
    }
}

// ---------------------------------------------------------------------------
// cls logit / is_electron / sd -> cholesky.  8 waves/block, 1 query per wave.
// ---------------------------------------------------------------------------
__global__ void head_kernel(const unsigned short* __restrict__ hCls,
                            const unsigned short* __restrict__ hSd,
                            const float* __restrict__ cls_w2,
                            const float* __restrict__ cls_b2,
                            const float* __restrict__ sd_w2,
                            const float* __restrict__ sd_b2,
                            float* __restrict__ out) {
    int t = threadIdx.x;
    int wave = t >> 5, lane = t & 31;
    int q = blockIdx.x * 8 + wave;
    if (q >= QTOT) return;

    const unsigned short* hc = hCls + (size_t)q * HID;
    float s = 0.f;
#pragma unroll 4
    for (int e = lane; e < HID; e += 32) s += bf16_to_f32(hc[e]) * cls_w2[e];
    s = wave_sum(s);
    float logit = s + cls_b2[0];

    const unsigned short* hs = hSd + (size_t)q * HID;
    float s0 = 0.f, s1 = 0.f, s2 = 0.f;
#pragma unroll 4
    for (int e = lane; e < HID; e += 32) {
        float h = bf16_to_f32(hs[e]);
        s0 += h * sd_w2[e * 3 + 0];
        s1 += h * sd_w2[e * 3 + 1];
        s2 += h * sd_w2[e * 3 + 2];
    }
    s0 = wave_sum(s0); s1 = wave_sum(s1); s2 = wave_sum(s2);

    if (lane == 0) {
        out[O_LOGIT + q] = logit;
        out[O_ISE + q] = (logit > 0.f) ? 1.f : 0.f;
        float d0 = expf(fminf(s0 + sd_b2[0], LOG_MAX_COV)) + EPS_F;
        float d1 = expf(fminf(s1 + sd_b2[1], LOG_MAX_COV)) + EPS_F;
        float od = s2 + sd_b2[2];
        size_t c = O_CHOL + (size_t)q * 4;
        out[c + 0] = d0; out[c + 1] = 0.f; out[c + 2] = od; out[c + 3] = d1;
    }
}

// ---------------------------------------------------------------------------
// Fused per-query window kernel: gather -> pm MLP (both layers, WMMA, in LDS)
// -> logits + index outputs.  1 block (256 threads) per query, 49->64 rows.
// LDS: X 64x128 bf16 (16K) + H 64x512 bf16 (64K) + Y 64x256 f32 (64K) + flags.
// Uses the 320KB/WGP CDNA5 LDS to keep the whole two-layer MLP on chip.
// ---------------------------------------------------------------------------
#define SMEM_X  0
#define SMEM_H  16384
#define SMEM_Y  (16384 + 65536)
#define SMEM_PL (16384 + 65536 + 65536)
#define SMEM_SZ (16384 + 65536 + 65536 + 512)

__global__ void fused_pm_kernel(const float* __restrict__ queries,
                                const int* __restrict__ indices,
                                const float* __restrict__ image,
                                const int* __restrict__ specified,
                                const unsigned short* __restrict__ W1P,
                                const float* __restrict__ b1,
                                const unsigned short* __restrict__ W2P,
                                const float* __restrict__ b2,
                                const float* __restrict__ mlpedQ,
                                float* __restrict__ out) {
    extern __shared__ unsigned char smem[];
    unsigned short* XS = (unsigned short*)(smem + SMEM_X);   // [64][128]
    unsigned short* HS = (unsigned short*)(smem + SMEM_H);   // [64][512]
    float* YS = (float*)(smem + SMEM_Y);                     // [64][256]
    float* PL = (float*)(smem + SMEM_PL);                    // [64] portion dots
    float* FB = PL + 64;                                     // [64] pad bias

    int t = threadIdx.x;
    int wave = t >> 5, lane = t & 31;
    int q = blockIdx.x;

    int bq = indices[q * 3 + 0];
    int y0 = indices[q * 3 + 1];
    int x0 = indices[q * 3 + 2];
    const float* mq = mlpedQ + (size_t)q * F_PIX;

    // ---- Phase 1: gather window, bf16-convert into XS, portion dot, flags ----
    for (int w = wave; w < NWIN; w += 8) {
        int wy = w / WS7, wx = w - wy * WS7;
        int ys = y0 + wy - 3, xs = x0 + wx - 3;
        bool vy = (ys >= 0) && (ys < HH);
        bool vx = (xs >= 0) && (xs < WW);
        int yc = min(max(ys, 0), HH - 1);
        int xc = min(max(xs, 0), WW - 1);
        size_t pix = ((size_t)bq * HH + yc) * WW + xc;
        const float* kp = image + pix * F_PIX;
        int c = lane * 4;
        float4 v = *(const float4*)(kp + c);
        XS[w * F_PIX + c + 0] = f32_to_bf16(v.x);
        XS[w * F_PIX + c + 1] = f32_to_bf16(v.y);
        XS[w * F_PIX + c + 2] = f32_to_bf16(v.z);
        XS[w * F_PIX + c + 3] = f32_to_bf16(v.w);
        float pl = v.x * mq[c] + v.y * mq[c + 1] + v.z * mq[c + 2] + v.w * mq[c + 3];
        pl = wave_sum(pl);
        if (lane == 0) {
            bool pad = !(vy && vx);
            bool isspec = (specified[pix] != 0) && !pad;
            PL[w] = pl;
            FB[w] = isspec ? 0.f : NEG_BIAS;
            out[O_ISPEC + (size_t)q * NWIN + w] = isspec ? 1.f : 0.f;
            size_t ki = O_KIDX + ((size_t)q * NWIN + w) * 3;
            out[ki + 0] = (float)bq; out[ki + 1] = (float)yc; out[ki + 2] = (float)xc;
            size_t si = O_SIDX + ((size_t)q * NWIN + w) * 4;
            out[si + 0] = (float)bq; out[si + 1] = (float)yc;
            out[si + 2] = (float)xc; out[si + 3] = (float)(q & (QPB - 1));
        }
    }
    // zero pad rows 49..63 of X (15*128 bf16 = 240 x 16B)
    {
        uint4 z; z.x = z.y = z.z = z.w = 0u;
        for (int i = t; i < 240; i += 256)
            ((uint4*)(XS + NWIN * F_PIX))[i] = z;
    }
    __syncthreads();

    int g = lane >> 4, ln = lane & 15;
    int wm = wave & 3;                 // row tile (0..3)

    // ---- Phase 2: layer 1  H[64,512] = relu(X[64,128] @ W1 + b1) ----
    {
        int cb = (wave >> 2) * 16;     // col-tile base: 0 or 16 (of 32)
#pragma unroll
        for (int pass = 0; pass < 2; ++pass) {
            v8f acc[8];
#pragma unroll
            for (int j = 0; j < 8; ++j) acc[j] = (v8f){0.f,0.f,0.f,0.f,0.f,0.f,0.f,0.f};
#pragma unroll
            for (int kt = 0; kt < 4; ++kt) {               // K = 128
                v16bf a = load_a_lds(XS, F_PIX, wm * 16, kt * 32, lane);
                v16bf bf[8];
#pragma unroll
                for (int j = 0; j < 8; ++j)
                    bf[j] = load_b_packed(W1P, HID >> 4, kt, cb + pass * 8 + j, lane);
#pragma unroll
                for (int j = 0; j < 8; ++j)
                    acc[j] = wmma_bf16(a, bf[j], acc[j]);
            }
#pragma unroll
            for (int j = 0; j < 8; ++j) {
                int col = (cb + pass * 8 + j) * 16 + ln;
                float bv = b1[col];
#pragma unroll
                for (int i = 0; i < 8; ++i) {
                    int row = wm * 16 + g * 8 + i;
                    HS[row * HID + col] = f32_to_bf16(fmaxf(acc[j][i] + bv, 0.f));
                }
            }
        }
    }
    __syncthreads();

    // ---- Phase 3: layer 2  Y[64,256] = H[64,512] @ W2 + b2 ----
    {
        int cb2 = (wave >> 2) * 8;     // col-tile base: 0 or 8 (of 16)
        v8f acc[8];
#pragma unroll
        for (int j = 0; j < 8; ++j) acc[j] = (v8f){0.f,0.f,0.f,0.f,0.f,0.f,0.f,0.f};
#pragma unroll 2
        for (int kt = 0; kt < 16; ++kt) {                  // K = 512
            v16bf a = load_a_lds(HS, HID, wm * 16, kt * 32, lane);
            v16bf bf[8];
#pragma unroll
            for (int j = 0; j < 8; ++j)
                bf[j] = load_b_packed(W2P, QD >> 4, kt, cb2 + j, lane);
#pragma unroll
            for (int j = 0; j < 8; ++j)
                acc[j] = wmma_bf16(a, bf[j], acc[j]);
        }
#pragma unroll
        for (int j = 0; j < 8; ++j) {
            int col = (cb2 + j) * 16 + ln;
            float bv = b2[col];
#pragma unroll
            for (int i = 0; i < 8; ++i) {
                int row = wm * 16 + g * 8 + i;
                YS[row * QD + col] = acc[j][i] + bv;
            }
        }
    }
    __syncthreads();

    // ---- Phase 4: binary_mask_logits = q . Y  (+bias); portion_logits out ----
    const float* qrow = queries + (size_t)q * QD;
    for (int r = wave; r < NWIN; r += 8) {
        float s = 0.f;
#pragma unroll
        for (int jj = 0; jj < 8; ++jj) {
            int c = lane + jj * 32;
            s += qrow[c] * YS[r * QD + c];
        }
        s = wave_sum(s);
        if (lane == 0) {
            float bias = FB[r];
            out[O_BML + (size_t)q * NWIN + r] = s + bias;
            out[O_PL  + (size_t)q * NWIN + r] = PL[r] + bias;
        }
    }
}

// ---------------------------------------------------------------------------
// Host-side launch
// ---------------------------------------------------------------------------
extern "C" void kernel_launch(void* const* d_in, const int* in_sizes, int n_in,
                              void* d_out, int out_size, void* d_ws, size_t ws_size,
                              hipStream_t stream) {
    (void)in_sizes; (void)n_in; (void)out_size; (void)ws_size;

    const float* queries   = (const float*)d_in[0];
    const int*   indices   = (const int*)d_in[1];
    const float* image     = (const float*)d_in[2];
    const int*   specified = (const int*)d_in[3];
    const float* cls_w1 = (const float*)d_in[4];
    const float* cls_b1 = (const float*)d_in[5];
    const float* cls_w2 = (const float*)d_in[6];
    const float* cls_b2 = (const float*)d_in[7];
    const float* pm_w1  = (const float*)d_in[8];
    const float* pm_b1  = (const float*)d_in[9];
    const float* pm_w2  = (const float*)d_in[10];
    const float* pm_b2  = (const float*)d_in[11];
    const float* qp_w1  = (const float*)d_in[12];
    const float* qp_b1  = (const float*)d_in[13];
    const float* qp_w2  = (const float*)d_in[14];
    const float* qp_b2  = (const float*)d_in[15];
    const float* sd_w1  = (const float*)d_in[16];
    const float* sd_b1  = (const float*)d_in[17];
    const float* sd_w2  = (const float*)d_in[18];
    const float* sd_b2  = (const float*)d_in[19];

    float* out = (float*)d_out;
    unsigned char* ws = (unsigned char*)d_ws;
    unsigned short* qB     = (unsigned short*)(ws + WS_QB);
    unsigned short* clsW1P = (unsigned short*)(ws + WS_CLSW1P);
    unsigned short* qpW1P  = (unsigned short*)(ws + WS_QPW1P);
    unsigned short* sdW1P  = (unsigned short*)(ws + WS_SDW1P);
    unsigned short* pmW1P  = (unsigned short*)(ws + WS_PMW1P);
    unsigned short* pmW2P  = (unsigned short*)(ws + WS_PMW2P);
    unsigned short* qpW2P  = (unsigned short*)(ws + WS_QPW2P);
    unsigned short* hCls   = (unsigned short*)(ws + WS_HCLS);
    unsigned short* hQp    = (unsigned short*)(ws + WS_HQP);
    unsigned short* hSd    = (unsigned short*)(ws + WS_HSD);
    float*          mlpedQ = (float*)(ws + WS_MLPQ);

    // 1) precision conversion / weight repack
    cvt_bf16_kernel<<<(QTOT * QD + 255) / 256, 256, 0, stream>>>(queries, qB, QTOT * QD);
    pack_b_kernel<<<(QD * HID + 255) / 256, 256, 0, stream>>>(cls_w1, clsW1P, QD, HID);
    pack_b_kernel<<<(QD * HID + 255) / 256, 256, 0, stream>>>(qp_w1,  qpW1P,  QD, HID);
    pack_b_kernel<<<(QD * HID + 255) / 256, 256, 0, stream>>>(sd_w1,  sdW1P,  QD, HID);
    pack_b_kernel<<<(F_PIX * HID + 255) / 256, 256, 0, stream>>>(pm_w1, pmW1P, F_PIX, HID);
    pack_b_kernel<<<(HID * QD + 255) / 256, 256, 0, stream>>>(pm_w2,  pmW2P,  HID, QD);
    pack_b_kernel<<<(HID * F_PIX + 255) / 256, 256, 0, stream>>>(qp_w2, qpW2P, HID, F_PIX);

    // 2) query-side layer-1 GEMMs:  [8192,256] x [256,512] -> relu bf16
    dim3 g1(QTOT / 128, HID / 128);
    gemm_wmma_kernel<<<g1, 256, 0, stream>>>(qB, clsW1P, cls_b1, hCls, QTOT, HID, QD, 1, 0);
    gemm_wmma_kernel<<<g1, 256, 0, stream>>>(qB, qpW1P,  qp_b1,  hQp,  QTOT, HID, QD, 1, 0);
    gemm_wmma_kernel<<<g1, 256, 0, stream>>>(qB, sdW1P,  sd_b1,  hSd,  QTOT, HID, QD, 1, 0);

    // 3) mlped_queries = hQp @ qp_w2 + qp_b2 : [8192,512] x [512,128] -> f32
    dim3 g2(QTOT / 128, F_PIX / 128);
    gemm_wmma_kernel<<<g2, 256, 0, stream>>>(hQp, qpW2P, qp_b2, mlpedQ, QTOT, F_PIX, HID, 0, 1);

    // 4) cls head + sd/cholesky
    head_kernel<<<QTOT / 8, 256, 0, stream>>>(hCls, hSd, cls_w2, cls_b2, sd_w2, sd_b2, out);

    // 5) fused gather + pm MLP + logits (dominant compute, WMMA bf16)
    fused_pm_kernel<<<QTOT, 256, SMEM_SZ, stream>>>(queries, indices, image, specified,
                                                    pmW1P, pm_b1, pmW2P, pm_b2,
                                                    mlpedQ, out);
}
